// BipolarMorphological2D_Smorph_20409684591420
// MI455X (gfx1250) — matrix operand order repaired and checked by reading, not compiled, not run on hardware
//
#include <hip/hip_runtime.h>

// ---------------------------------------------------------------------------
// BipolarMorphological2D / SMorph, refactored to WMMA GEMMs for gfx1250.
//
//   m_p = max(+-x_patch, 0.1),  L_p = m_p * ln(m_p)
//   E_po = exp(k_po),           F_po = k_po * exp(k_po)
//   D = M @ E,  N = L @ E + M @ F,  branch = exp(N / D)
//   out = y11 - y12 - y21 + y22 + bias
//
// Shapes: B=4, C=32, H=W=32, O=64, K=3 -> rows = 4*30*30 = 3600 (225 tiles
// of 16), reduction P = 9*32 = 288 (9 WMMA K=32 steps), O = 64 (4 N tiles).
// All dimensions divide exactly; no padding required anywhere.
// ---------------------------------------------------------------------------

typedef __attribute__((ext_vector_type(16))) _Float16 v16h;
typedef __attribute__((ext_vector_type(8)))  float    v8f;

#define B_   4
#define C_   32
#define H_   32
#define W_   32
#define O_   64
#define KW_  3
#define HO_  30
#define WO_  30
#define P_   288            // KW*KW*C
#define ROWS 3600           // B*HO*WO
#define RT_  225            // ROWS / 16 row tiles
#define KB_  9              // P / 32 K-blocks
#define OT_  4              // O / 16 col tiles
#define FRAG 512            // halves per fragment: 32 lanes * 16 halves

// ---------------------------------------------------------------------------
// Stage 1: patches -> M1/L1/M2/L2 in fragment-ready f16 layout.
// Linear tid == storage offset (fully coalesced stores). A-matrix 16x32 f16
// VGPR layout (ISA 7.12.2): lane&15 = row M; halves 0..7 -> K base 0 (lanes
// 0-15) or 8 (lanes 16-31); halves 8..15 -> +16.
// ---------------------------------------------------------------------------
__global__ void prep_A_kernel(const float* __restrict__ x,
                              _Float16* __restrict__ M1, _Float16* __restrict__ L1,
                              _Float16* __restrict__ M2, _Float16* __restrict__ L2) {
  int tid = blockIdx.x * blockDim.x + threadIdx.x;
  if (tid >= RT_ * KB_ * FRAG) return;
  int h    = tid & 15;
  int lane = (tid >> 4) & 31;
  int blk  = tid >> 9;               // rt*KB_ + kb
  int kb   = blk % KB_;
  int rt   = blk / KB_;

  int row = rt * 16 + (lane & 15);
  int K   = kb * 32 + ((h & 8) ? 16 : 0) + ((lane & 16) ? 8 : 0) + (h & 7);

  // row -> (b, hh, ww); K -> (kh, kw, c), patch ordering p = (kh*3+kw)*32+c
  int b    = row / (HO_ * WO_);
  int r2   = row - b * (HO_ * WO_);
  int hh   = r2 / WO_;
  int ww   = r2 - hh * WO_;
  int c    = K & 31;
  int khkw = K >> 5;
  int kh   = khkw / KW_;
  int kw   = khkw - kh * KW_;

  float xv = x[((b * C_ + c) * H_ + (hh + kh)) * W_ + (ww + kw)];
  float m1 = fmaxf(xv, 0.1f);
  float m2 = fmaxf(-xv, 0.1f);
  M1[tid] = (_Float16)m1;
  L1[tid] = (_Float16)(m1 * __logf(m1));
  M2[tid] = (_Float16)m2;
  L2[tid] = (_Float16)(m2 * __logf(m2));
}

// ---------------------------------------------------------------------------
// Stage 2: weights -> E1/F1/E2/F2 in B-fragment order.
// B-matrix 32x16 f16 layout: lane&15 = col N; halves 0..7 -> K base 0 (lanes
// 0-15) or 8 (lanes 16-31); halves 8..15 -> +16 (same order as A).
// ---------------------------------------------------------------------------
__global__ void prep_B_kernel(const float* __restrict__ k1, const float* __restrict__ k2,
                              _Float16* __restrict__ E1, _Float16* __restrict__ F1,
                              _Float16* __restrict__ E2, _Float16* __restrict__ F2) {
  int tid = blockIdx.x * blockDim.x + threadIdx.x;
  if (tid >= OT_ * KB_ * FRAG) return;
  int h    = tid & 15;
  int lane = (tid >> 4) & 31;
  int blk  = tid >> 9;               // ot*KB_ + kb
  int kb   = blk % KB_;
  int ot   = blk / KB_;

  int K = kb * 32 + ((h & 8) ? 16 : 0) + ((lane & 16) ? 8 : 0) + (h & 7);
  int o = ot * 16 + (lane & 15);

  float kv1 = k1[K * O_ + o];        // k.reshape(288, 64) is row-major
  float kv2 = k2[K * O_ + o];
  float e1  = __expf(kv1);
  float e2  = __expf(kv2);
  E1[tid] = (_Float16)e1;
  F1[tid] = (_Float16)(kv1 * e1);
  E2[tid] = (_Float16)e2;
  F2[tid] = (_Float16)(kv2 * e2);
}

// ---------------------------------------------------------------------------
// Stage 3: one wave per 16x16 output tile. 9 K-steps x 12 WMMAs.
// Accumulators: D_ij = M_i@E_j ; N_ij = L_i@E_j + M_i@F_j  (i=input sign,
// j=kernel). Epilogue: exp(N/D) with signs +,-,-,+ and bias.
// ---------------------------------------------------------------------------
__global__ void __launch_bounds__(32)
smorph_gemm_kernel(const _Float16* __restrict__ M1, const _Float16* __restrict__ L1,
                   const _Float16* __restrict__ M2, const _Float16* __restrict__ L2,
                   const _Float16* __restrict__ E1, const _Float16* __restrict__ F1,
                   const _Float16* __restrict__ E2, const _Float16* __restrict__ F2,
                   const float* __restrict__ bias, float* __restrict__ out) {
  int tile = blockIdx.x;             // 0 .. RT_*OT_-1
  int rt   = tile >> 2;
  int ot   = tile & 3;
  int lane = threadIdx.x;

  v8f d11 = {}, d12 = {}, d21 = {}, d22 = {};
  v8f n11 = {}, n12 = {}, n21 = {}, n22 = {};

  size_t aoff = (size_t)(rt * KB_) * FRAG + (size_t)lane * 16;
  size_t boff = (size_t)(ot * KB_) * FRAG + (size_t)lane * 16;

#pragma unroll
  for (int kb = 0; kb < KB_; ++kb) {
    size_t ao = aoff + (size_t)kb * FRAG;
    size_t bo = boff + (size_t)kb * FRAG;
    v16h aM1 = *(const v16h*)(M1 + ao);
    v16h aL1 = *(const v16h*)(L1 + ao);
    v16h aM2 = *(const v16h*)(M2 + ao);
    v16h aL2 = *(const v16h*)(L2 + ao);
    v16h bE1 = *(const v16h*)(E1 + bo);
    v16h bF1 = *(const v16h*)(F1 + bo);
    v16h bE2 = *(const v16h*)(E2 + bo);
    v16h bF2 = *(const v16h*)(F2 + bo);

    d11 = __builtin_amdgcn_wmma_f32_16x16x32_f16(false, aM1, false, bE1, (short)0, d11, false, false);
    d12 = __builtin_amdgcn_wmma_f32_16x16x32_f16(false, aM1, false, bE2, (short)0, d12, false, false);
    d21 = __builtin_amdgcn_wmma_f32_16x16x32_f16(false, aM2, false, bE1, (short)0, d21, false, false);
    d22 = __builtin_amdgcn_wmma_f32_16x16x32_f16(false, aM2, false, bE2, (short)0, d22, false, false);

    n11 = __builtin_amdgcn_wmma_f32_16x16x32_f16(false, aL1, false, bE1, (short)0, n11, false, false);
    n11 = __builtin_amdgcn_wmma_f32_16x16x32_f16(false, aM1, false, bF1, (short)0, n11, false, false);
    n12 = __builtin_amdgcn_wmma_f32_16x16x32_f16(false, aL1, false, bE2, (short)0, n12, false, false);
    n12 = __builtin_amdgcn_wmma_f32_16x16x32_f16(false, aM1, false, bF2, (short)0, n12, false, false);
    n21 = __builtin_amdgcn_wmma_f32_16x16x32_f16(false, aL2, false, bE1, (short)0, n21, false, false);
    n21 = __builtin_amdgcn_wmma_f32_16x16x32_f16(false, aM2, false, bF1, (short)0, n21, false, false);
    n22 = __builtin_amdgcn_wmma_f32_16x16x32_f16(false, aL2, false, bE2, (short)0, n22, false, false);
    n22 = __builtin_amdgcn_wmma_f32_16x16x32_f16(false, aM2, false, bF2, (short)0, n22, false, false);
  }

  // C/D layout: VGPR v, lanes 0-15 -> M=v ; lanes 16-31 -> M=v+8 ; N=lane&15
  int o     = ot * 16 + (lane & 15);
  float bi  = bias[o];
  int mbase = (lane >> 4) * 8;

#pragma unroll
  for (int v = 0; v < 8; ++v) {
    float r = __expf(n11[v] / d11[v]) - __expf(n12[v] / d12[v])
            - __expf(n21[v] / d21[v]) + __expf(n22[v] / d22[v]) + bi;
    int row = rt * 16 + mbase + v;
    int b   = row / (HO_ * WO_);
    int r2  = row - b * (HO_ * WO_);
    int hh  = r2 / WO_;
    int ww  = r2 - hh * WO_;
    out[((b * O_ + o) * HO_ + hh) * WO_ + ww] = r;
  }
}

// ---------------------------------------------------------------------------
extern "C" void kernel_launch(void* const* d_in, const int* in_sizes, int n_in,
                              void* d_out, int out_size, void* d_ws, size_t ws_size,
                              hipStream_t stream) {
  const float* x    = (const float*)d_in[0];
  const float* k1   = (const float*)d_in[1];
  const float* k2   = (const float*)d_in[2];
  const float* bias = (const float*)d_in[3];
  float*       out  = (float*)d_out;

  _Float16* ws = (_Float16*)d_ws;
  const size_t A_per = (size_t)RT_ * KB_ * FRAG;   // 1,036,800 halves / matrix
  const size_t B_per = (size_t)OT_ * KB_ * FRAG;   //    18,432 halves / matrix
  _Float16* M1 = ws;
  _Float16* L1 = ws + 1 * A_per;
  _Float16* M2 = ws + 2 * A_per;
  _Float16* L2 = ws + 3 * A_per;
  _Float16* E1 = ws + 4 * A_per;
  _Float16* F1 = E1 + 1 * B_per;
  _Float16* E2 = E1 + 2 * B_per;
  _Float16* F2 = E1 + 3 * B_per;
  // total ws use: 4*A_per + 4*B_per halves = ~8.05 MB

  prep_A_kernel<<<(RT_ * KB_ * FRAG) / 256, 256, 0, stream>>>(x, M1, L1, M2, L2);
  prep_B_kernel<<<(OT_ * KB_ * FRAG) / 256, 256, 0, stream>>>(k1, k2, E1, F1, E2, F2);
  smorph_gemm_kernel<<<RT_ * OT_, 32, 0, stream>>>(M1, L1, M2, L2, E1, F1, E2, F2, bias, out);
}
